// CrossWindowAttention3D_29343216566563
// MI455X (gfx1250) — matrix operand order repaired and checked by reading, not compiled, not run on hardware
//
#include <hip/hip_runtime.h>
#include <hip/hip_bf16.h>

typedef __attribute__((ext_vector_type(16))) _Float16 v16h;
typedef __attribute__((ext_vector_type(8)))  _Float16 v8h;
typedef __attribute__((ext_vector_type(8)))  float    v8f;

constexpr int TOK   = 98;    // tokens per window
constexpr int TOKP  = 128;   // tokens padded (multiple of 32 for K-dim of P@V)
constexpr int CH    = 96;    // embed dim
constexpr int NHEAD = 3;
constexpr int HD    = 32;    // head dim (== WMMA K for f16!)
constexpr int NWIN  = 512;
constexpr int NB    = 1024;  // B_

#define WMMA_F16(A, B, Cacc) \
  __builtin_amdgcn_wmma_f32_16x16x32_f16(false, (A), false, (B), (short)0, (Cacc), false, false)

// A operand (16x32 f16): lane holds row m0+(lane&15); halves 0..7 = K (lane/16)*8+0..7,
// halves 8..15 = K (lane/16)*8+16..23  -> two contiguous 16B chunks from row-major storage.
static __device__ __forceinline__ v16h frag_a(const _Float16* base, int pitch,
                                              int m0, int k0, int lane) {
  const _Float16* p = base + (size_t)(m0 + (lane & 15)) * pitch + k0 + ((lane >> 4) << 3);
  v8h lo = *(const v8h*)(p);
  v8h hi = *(const v8h*)(p + 16);
  v16h a;
#pragma unroll
  for (int i = 0; i < 8; ++i) { a[i] = lo[i]; a[i + 8] = hi[i]; }
  return a;
}

// B operand (32x16 f16): lane holds column n0+(lane&15); halves 0..7 = K (lane/16)*8+0..7,
// halves 8..15 = K (lane/16)*8+16..23.  bt is B^T stored row-major (K-contiguous per column).
static __device__ __forceinline__ v16h frag_b(const _Float16* bt, int pitch,
                                              int n0, int k0, int lane) {
  const _Float16* p = bt + (size_t)(n0 + (lane & 15)) * pitch + k0 + ((lane >> 4) << 3);
  v8h lo = *(const v8h*)(p);
  v8h hi = *(const v8h*)(p + 16);
  v16h b;
#pragma unroll
  for (int i = 0; i < 8; ++i) { b[i] = lo[i]; b[i + 8] = hi[i]; }
  return b;
}

__global__ void convert_weights_kernel(const float* __restrict__ qkv_w,
                                       const float* __restrict__ proj_w,
                                       _Float16* __restrict__ wh,
                                       _Float16* __restrict__ pwh) {
  int i = blockIdx.x * blockDim.x + threadIdx.x;
  if (i < 3 * CH * CH) wh[i]  = (_Float16)qkv_w[i];
  if (i < CH * CH)     pwh[i] = (_Float16)proj_w[i];
}

__global__ void __launch_bounds__(256)
win_attn_kernel(const float* __restrict__ x, const float* __restrict__ y,
                const float* __restrict__ mask, const _Float16* __restrict__ wh,
                const float* __restrict__ rpb, const _Float16* __restrict__ pwh,
                const float* __restrict__ proj_b, float* __restrict__ out) {
  extern __shared__ char smem[];
  _Float16* sA  = (_Float16*)smem;        // 128x96  staged x / y (f16)
  _Float16* sK  = sA  + TOKP * CH;        // 128x96  K   (row-major, token x hd*3)
  _Float16* sQ  = sK  + TOKP * CH;        // 128x96  Q   (pre-scaled)
  _Float16* sVT = sQ  + TOKP * CH;        // 96x128  V^T (head-dim x token)
  _Float16* sO  = sVT + CH * TOKP;        // 128x96  attention output (f16)
  _Float16* sP  = sO  + TOKP * CH;        // 128x128 softmax probs, per-wave strips

  const int b    = blockIdx.x;
  const int tid  = threadIdx.x;
  const int w    = tid >> 5;
  const int lane = tid & 31;
  const int lm   = (lane >> 4) << 3;      // D-fragment row group (0 or 8)
  const int ln   = lane & 15;             // D-fragment column within tile
  const int wwin = b & (NWIN - 1);        // mask row = b % 512

  // ---- Phase A: x -> sA (f32 -> f16, zero-pad rows 98..127) -----------------
  for (int i = tid; i < TOKP * CH; i += 256) {
    int m = i / CH;
    sA[i] = (_Float16)((m < TOK) ? x[(size_t)b * TOK * CH + i] : 0.f);
  }
  __syncthreads();

  // ---- GEMM1: K = xW_k^T, V^T = (xW_v^T)^T  (96 tile jobs over 8 waves) -----
  for (int job = w; job < 96; job += 8) {
    int isV = (job >= 48) ? 1 : 0;
    int j   = job - isV * 48;
    int mt  = j / 6, nt = j % 6;
    const _Float16* bw = wh + (size_t)(CH + isV * CH) * CH;  // k rows 96.., v rows 192..
    v8f acc = {};
#pragma unroll
    for (int kk = 0; kk < 3; ++kk) {
      v16h a  = frag_a(sA, CH, mt * 16, kk * 32, lane);
      v16h bb = frag_b(bw, CH, nt * 16, kk * 32, lane);
      acc = WMMA_F16(a, bb, acc);
    }
    if (!isV) {
#pragma unroll
      for (int r = 0; r < 8; ++r)
        sK[(size_t)(mt * 16 + lm + r) * CH + nt * 16 + ln] = (_Float16)acc[r];
    } else {  // store transposed: sVT[head-dim][token]
#pragma unroll
      for (int r = 0; r < 8; ++r)
        sVT[(size_t)(nt * 16 + ln) * TOKP + mt * 16 + lm + r] = (_Float16)acc[r];
    }
  }
  __syncthreads();

  // ---- Phase C: y -> sA -----------------------------------------------------
  for (int i = tid; i < TOKP * CH; i += 256) {
    int m = i / CH;
    sA[i] = (_Float16)((m < TOK) ? y[(size_t)b * TOK * CH + i] : 0.f);
  }
  __syncthreads();

  // ---- GEMM2: Q = (yW_q^T) * hd^-0.5 ---------------------------------------
  const float scale = 0.17677669529663687f;  // 32^-0.5
  for (int job = w; job < 48; job += 8) {
    int mt = job / 6, nt = job % 6;
    v8f acc = {};
#pragma unroll
    for (int kk = 0; kk < 3; ++kk) {
      v16h a  = frag_a(sA, CH, mt * 16, kk * 32, lane);
      v16h bb = frag_b(wh, CH, nt * 16, kk * 32, lane);
      acc = WMMA_F16(a, bb, acc);
    }
#pragma unroll
    for (int r = 0; r < 8; ++r)
      sQ[(size_t)(mt * 16 + lm + r) * CH + nt * 16 + ln] = (_Float16)(acc[r] * scale);
  }
  __syncthreads();

  // ---- Phase D: attention; wave w owns token rows [16w, 16w+16) -------------
  const int m0 = w * 16;
  _Float16* sPw = sP + w * 16 * TOKP;
  for (int h = 0; h < NHEAD; ++h) {
    v8f s[7];
    v16h aq = frag_a(sQ, CH, m0, h * HD, lane);   // K = hd = 32 -> single WMMA per tile
#pragma unroll
    for (int nt = 0; nt < 7; ++nt) {
      v16h bk = frag_b(sK, CH, nt * 16, h * HD, lane);
      v8f z = {};
      s[nt] = WMMA_F16(aq, bk, z);
    }
    // relative position bias (computed inline from (2,7,7) geometry) + mask
#pragma unroll
    for (int nt = 0; nt < 7; ++nt) {
      int n = nt * 16 + ln;
      if (n < TOK) {
        int zj = n / 49, rj = n % 49, yj = rj / 7, xj = rj % 7;
#pragma unroll
        for (int r = 0; r < 8; ++r) {
          int m = m0 + lm + r; if (m > TOK - 1) m = TOK - 1;  // pad rows: clamp (unused)
          int zi = m / 49, ri = m % 49, yi = ri / 7, xi = ri % 7;
          int idx = (zi - zj + 1) * 169 + (yi - yj + 6) * 13 + (xi - xj + 6);
          s[nt][r] += rpb[idx * NHEAD + h] +
                      mask[(size_t)wwin * TOK * TOK + (size_t)m * TOK + n];
        }
      } else {
#pragma unroll
        for (int r = 0; r < 8; ++r) s[nt][r] = -1e30f;   // pad cols -> exp() == 0
      }
    }
    // softmax per row: reduce across the 16-lane half holding that row
#pragma unroll
    for (int r = 0; r < 8; ++r) {
      float mx = -1e30f;
#pragma unroll
      for (int nt = 0; nt < 7; ++nt) mx = fmaxf(mx, s[nt][r]);
#pragma unroll
      for (int off = 8; off >= 1; off >>= 1) mx = fmaxf(mx, __shfl_xor(mx, off, 32));
      float sum = 0.f;
#pragma unroll
      for (int nt = 0; nt < 7; ++nt) { float e = __expf(s[nt][r] - mx); s[nt][r] = e; sum += e; }
#pragma unroll
      for (int off = 8; off >= 1; off >>= 1) sum += __shfl_xor(sum, off, 32);
      float inv = 1.f / sum;
#pragma unroll
      for (int nt = 0; nt < 7; ++nt) s[nt][r] *= inv;
    }
    // bounce P through wave-private LDS strip (D-layout -> A-layout), zero K-pad 112..127
#pragma unroll
    for (int nt = 0; nt < 7; ++nt)
#pragma unroll
      for (int r = 0; r < 8; ++r)
        sPw[(lm + r) * TOKP + nt * 16 + ln] = (_Float16)s[nt][r];
#pragma unroll
    for (int r = 0; r < 8; ++r)
      sPw[(lm + r) * TOKP + 112 + ln] = (_Float16)0.f;

    // P @ V : K = 128 tokens (4 WMMAs), N = 32 head dims (2 tiles)
#pragma unroll
    for (int n2 = 0; n2 < 2; ++n2) {
      v8f acc = {};
#pragma unroll
      for (int kk = 0; kk < 4; ++kk) {
        v16h a  = frag_a(sPw, TOKP, 0, kk * 32, lane);
        v16h bv = frag_b(sVT, TOKP, h * HD + n2 * 16, kk * 32, lane);
        acc = WMMA_F16(a, bv, acc);
      }
#pragma unroll
      for (int r = 0; r < 8; ++r)
        sO[(size_t)(m0 + lm + r) * CH + h * HD + n2 * 16 + ln] = (_Float16)acc[r];
    }
  }
  __syncthreads();

  // ---- Phase E: out = sO @ proj_w^T + proj_b (f32 stores) -------------------
  for (int job = w; job < 48; job += 8) {
    int mt = job / 6, nt = job % 6;
    v8f acc = {};
#pragma unroll
    for (int kk = 0; kk < 3; ++kk) {
      v16h a  = frag_a(sO, CH, mt * 16, kk * 32, lane);
      v16h bb = frag_b(pwh, CH, nt * 16, kk * 32, lane);
      acc = WMMA_F16(a, bb, acc);
    }
    int n = nt * 16 + ln;
    float pb = proj_b[n];
#pragma unroll
    for (int r = 0; r < 8; ++r) {
      int m = mt * 16 + lm + r;
      if (m < TOK) out[(size_t)b * TOK * CH + (size_t)m * CH + n] = acc[r] + pb;
    }
  }
}

extern "C" void kernel_launch(void* const* d_in, const int* in_sizes, int n_in,
                              void* d_out, int out_size, void* d_ws, size_t ws_size,
                              hipStream_t stream) {
  (void)in_sizes; (void)n_in; (void)out_size; (void)ws_size;
  const float* x      = (const float*)d_in[0];
  const float* y      = (const float*)d_in[1];
  const float* mask   = (const float*)d_in[2];
  const float* qkv_w  = (const float*)d_in[3];
  const float* rpb    = (const float*)d_in[4];
  const float* proj_w = (const float*)d_in[5];
  const float* proj_b = (const float*)d_in[6];

  _Float16* wh  = (_Float16*)d_ws;          // 288x96 f16 qkv weights
  _Float16* pwh = wh + 3 * CH * CH;         // 96x96  f16 proj weights

  convert_weights_kernel<<<(3 * CH * CH + 255) / 256, 256, 0, stream>>>(qkv_w, proj_w, wh, pwh);

  // LDS: 5 * (128*96) + 128*128 halves = 152 KB dynamic
  size_t smem_bytes = (size_t)(5 * TOKP * CH + TOKP * TOKP) * sizeof(_Float16);
  win_attn_kernel<<<NB, 256, smem_bytes, stream>>>(x, y, mask, wh, rpb, pwh, proj_b,
                                                   (float*)d_out);
}
